// UCCModel_85581518340396
// MI455X (gfx1250) — compile-verified, matrix operand order checked by reading, not run on hardware
//
#include <hip/hip_runtime.h>
#include <hip/hip_bf16.h>

typedef _Float16 v16h __attribute__((ext_vector_type(16)));
typedef _Float16 h8   __attribute__((ext_vector_type(8)));
typedef float    v8f  __attribute__((ext_vector_type(8)));

#define CDIV(a,b) (((a)+(b)-1)/(b))

// ---------------------------------------------------------------------------
// Cast f32 [rows,K] -> f16 [Rp,Kp] with zero padding in both dims.
// Used for activations (Rp==rows) and weights (rows padded to mult of 64).
// ---------------------------------------------------------------------------
__global__ void cast_pad_f16_kernel(const float* __restrict__ src,
                                    _Float16* __restrict__ dst,
                                    int rows, int K, int Rp, int Kp) {
    long long i = (long long)blockIdx.x * blockDim.x + threadIdx.x;
    long long total = (long long)Rp * Kp;
    if (i >= total) return;
    int c = (int)(i % Kp);
    int r = (int)(i / Kp);
    dst[i] = (r < rows && c < K) ? (_Float16)src[(long long)r * K + c]
                                 : (_Float16)0.0f;
}

// ---------------------------------------------------------------------------
// WMMA GEMM: C[M,N] = A[M,Kp] * W[Np,Kp]^T + bias, optional ReLU.
// A: f16 row-major, K padded to Kp (mult of 32).
// W: f16 [Np,Kp], torch [out,in] layout padded to Np (mult of 64) rows —
// so ALL fragment loads are unconditional aligned 16B vectors.
// One wave32 computes a 16x64 tile: one A fragment feeds 4 WMMAs per K-step
// (4x A reuse, 4 independent accumulators for the scheduler to pipeline).
// Fragment layouts per CDNA5 ISA 7.12.2 (wave32):
//   A: lane holds row m=lane&15; halfs 0..7 = k in [kb,kb+8),
//      halfs 8..15 = k in [kb+16,kb+24), kb = (lane>=16)?8:0
//   B: lane holds col n=lane&15; same K mapping
//   C: elem r -> m = r + 8*(lane>=16), n = lane&15
// ---------------------------------------------------------------------------
__global__ void gemm_wmma_f16_kernel(const _Float16* __restrict__ A,
                                     const _Float16* __restrict__ W,
                                     const float* __restrict__ bias,
                                     float* __restrict__ C,
                                     int M, int N, int Np, int Kp, int relu) {
    int wave = threadIdx.x >> 5;
    int lane = threadIdx.x & 31;
    int tilesN = Np >> 6;                      // 16x64 tiles along N
    int tilesM = M >> 4;                       // M always a multiple of 16 here
    int tile = blockIdx.x * (blockDim.x >> 5) + wave;
    if (tile >= tilesM * tilesN) return;       // wave-uniform exit (EXEC full)

    int tm  = tile / tilesN;
    int tn4 = tile - tm * tilesN;
    int l15  = lane & 15;
    int hsel = lane >> 4;                      // 0 = lanes 0..15, 1 = 16..31
    int kb   = hsel * 8;

    const _Float16* ap = A + (long long)(tm * 16 + l15) * Kp;
    const _Float16* bp0 = W + (long long)(tn4 * 64 +  0 + l15) * Kp;
    const _Float16* bp1 = W + (long long)(tn4 * 64 + 16 + l15) * Kp;
    const _Float16* bp2 = W + (long long)(tn4 * 64 + 32 + l15) * Kp;
    const _Float16* bp3 = W + (long long)(tn4 * 64 + 48 + l15) * Kp;

    v8f acc0 = {0.f,0.f,0.f,0.f,0.f,0.f,0.f,0.f};
    v8f acc1 = acc0, acc2 = acc0, acc3 = acc0;

    for (int kk = 0; kk < Kp; kk += 32) {
        v16h a, b0, b1, b2, b3;
        h8 al = *(const h8*)(ap + kk + kb);
        h8 ah = *(const h8*)(ap + kk + kb + 16);
        h8 b0l = *(const h8*)(bp0 + kk + kb);
        h8 b0h = *(const h8*)(bp0 + kk + kb + 16);
        h8 b1l = *(const h8*)(bp1 + kk + kb);
        h8 b1h = *(const h8*)(bp1 + kk + kb + 16);
        h8 b2l = *(const h8*)(bp2 + kk + kb);
        h8 b2h = *(const h8*)(bp2 + kk + kb + 16);
        h8 b3l = *(const h8*)(bp3 + kk + kb);
        h8 b3h = *(const h8*)(bp3 + kk + kb + 16);
        #pragma unroll
        for (int j = 0; j < 8; ++j) {
            a[j]  = al[j];  a[8+j]  = ah[j];
            b0[j] = b0l[j]; b0[8+j] = b0h[j];
            b1[j] = b1l[j]; b1[8+j] = b1h[j];
            b2[j] = b2l[j]; b2[8+j] = b2h[j];
            b3[j] = b3l[j]; b3[8+j] = b3h[j];
        }
        acc0 = __builtin_amdgcn_wmma_f32_16x16x32_f16(false, a, false, b0, (short)0, acc0, false, false);
        acc1 = __builtin_amdgcn_wmma_f32_16x16x32_f16(false, a, false, b1, (short)0, acc1, false, false);
        acc2 = __builtin_amdgcn_wmma_f32_16x16x32_f16(false, a, false, b2, (short)0, acc2, false, false);
        acc3 = __builtin_amdgcn_wmma_f32_16x16x32_f16(false, a, false, b3, (short)0, acc3, false, false);
    }

    // epilogue: bias + ReLU + guarded store (N tail only here)
    #pragma unroll
    for (int j = 0; j < 4; ++j) {
        v8f acc = (j == 0) ? acc0 : (j == 1) ? acc1 : (j == 2) ? acc2 : acc3;
        int n = tn4 * 64 + j * 16 + l15;
        if (n < N) {
            float bv = bias[n];
            #pragma unroll
            for (int r = 0; r < 8; ++r) {
                int m = tm * 16 + r + hsel * 8;
                float v = acc[r] + bv;
                if (relu) v = fmaxf(v, 0.0f);
                C[(long long)m * N + n] = v;
            }
        }
    }
}

// ---------------------------------------------------------------------------
// Direct 3x3 conv, pad=1, stride s, optional ReLU.  w: [Cout,Cin,3,3]
// ---------------------------------------------------------------------------
__global__ void conv3x3_kernel(const float* __restrict__ in,
                               const float* __restrict__ w,
                               const float* __restrict__ bias,
                               float* __restrict__ out,
                               int Nb, int Cin, int Hin, int Win,
                               int Cout, int Hout, int Wout, int stride, int relu) {
    long long idx = (long long)blockIdx.x * blockDim.x + threadIdx.x;
    long long total = (long long)Nb * Cout * Hout * Wout;
    if (idx >= total) return;
    int wo = (int)(idx % Wout); long long t = idx / Wout;
    int ho = (int)(t % Hout);   t /= Hout;
    int co = (int)(t % Cout);   int n = (int)(t / Cout);

    float acc = bias[co];
    for (int ci = 0; ci < Cin; ++ci) {
        const float* ip = in + (((long long)n * Cin + ci) * Hin) * Win;
        const float* wp = w + ((co * Cin + ci) * 9);
        #pragma unroll
        for (int kh = 0; kh < 3; ++kh) {
            int hi = ho * stride - 1 + kh;
            if (hi < 0 || hi >= Hin) continue;
            #pragma unroll
            for (int kw = 0; kw < 3; ++kw) {
                int wi = wo * stride - 1 + kw;
                if (wi < 0 || wi >= Win) continue;
                acc += wp[kh * 3 + kw] * ip[(long long)hi * Win + wi];
            }
        }
    }
    if (relu) acc = fmaxf(acc, 0.0f);
    out[idx] = acc;
}

// ---------------------------------------------------------------------------
// ConvTranspose2d k=3, pad=1, output_pad=stride-1 (gather form).
// w: torch layout [Cin,Cout,3,3].  Hout = Hin*stride.
// ---------------------------------------------------------------------------
__global__ void convT3x3_kernel(const float* __restrict__ in,
                                const float* __restrict__ w,
                                const float* __restrict__ bias,
                                float* __restrict__ out,
                                int Nb, int Cin, int Hin, int Win,
                                int Cout, int Hout, int Wout, int stride, int relu) {
    long long idx = (long long)blockIdx.x * blockDim.x + threadIdx.x;
    long long total = (long long)Nb * Cout * Hout * Wout;
    if (idx >= total) return;
    int wo = (int)(idx % Wout); long long t = idx / Wout;
    int ho = (int)(t % Hout);   t /= Hout;
    int co = (int)(t % Cout);   int n = (int)(t / Cout);

    float acc = bias[co];
    #pragma unroll
    for (int kh = 0; kh < 3; ++kh) {
        int th = ho + 1 - kh;
        if (th < 0 || (th % stride) != 0) continue;
        int hi = th / stride;
        if (hi >= Hin) continue;
        #pragma unroll
        for (int kw = 0; kw < 3; ++kw) {
            int tw = wo + 1 - kw;
            if (tw < 0 || (tw % stride) != 0) continue;
            int wi = tw / stride;
            if (wi >= Win) continue;
            for (int ci = 0; ci < Cin; ++ci) {
                float iv = in[(((long long)n * Cin + ci) * Hin + hi) * Win + wi];
                float wv = w[((ci * Cout + co) * 3 + kh) * 3 + kw];
                acc += wv * iv;
            }
        }
    }
    if (relu) acc = fmaxf(acc, 0.0f);
    out[idx] = acc;
}

// ---------------------------------------------------------------------------
// KDE over instances: feat [B*I, F] -> fd [B, F*11].  The Gaussian prefactor
// alpha cancels in the per-(b,f) normalization; beta = -1/(2*0.1^2) = -50.
// ---------------------------------------------------------------------------
__global__ void kde_kernel(const float* __restrict__ feat,
                           float* __restrict__ fd, int B, int I, int F) {
    int idx = blockIdx.x * blockDim.x + threadIdx.x;
    if (idx >= B * F) return;
    int f = idx % F;
    int b = idx / F;
    float s[11];
    #pragma unroll
    for (int k = 0; k < 11; ++k) s[k] = 0.0f;
    for (int i = 0; i < I; ++i) {
        float v = feat[((long long)b * I + i) * F + f];
        #pragma unroll
        for (int k = 0; k < 11; ++k) {
            float d = 0.1f * (float)k - v;
            s[k] += __expf(-50.0f * d * d);
        }
    }
    float tot = 0.0f;
    #pragma unroll
    for (int k = 0; k < 11; ++k) tot += s[k];
    float inv = 1.0f / tot;
    #pragma unroll
    for (int k = 0; k < 11; ++k)
        fd[(long long)b * (F * 11) + f * 11 + k] = s[k] * inv;
}

// ---------------------------------------------------------------------------
// Row softmax for [B,4] logits.
// ---------------------------------------------------------------------------
__global__ void softmax4_kernel(const float* __restrict__ logits,
                                float* __restrict__ out, int B) {
    int b = blockIdx.x * blockDim.x + threadIdx.x;
    if (b >= B) return;
    float v0 = logits[b*4+0], v1 = logits[b*4+1], v2 = logits[b*4+2], v3 = logits[b*4+3];
    float m = fmaxf(fmaxf(v0, v1), fmaxf(v2, v3));
    float e0 = __expf(v0-m), e1 = __expf(v1-m), e2 = __expf(v2-m), e3 = __expf(v3-m);
    float inv = 1.0f / (e0+e1+e2+e3);
    out[b*4+0] = e0*inv; out[b*4+1] = e1*inv; out[b*4+2] = e2*inv; out[b*4+3] = e3*inv;
}

// ---------------------------------------------------------------------------

extern "C" void kernel_launch(void* const* d_in, const int* in_sizes, int n_in,
                              void* d_out, int out_size, void* d_ws, size_t ws_size,
                              hipStream_t stream) {
    (void)in_sizes; (void)n_in; (void)out_size; (void)ws_size;

    const int BI = 2048;        // 64 bags * 32 instances
    const int B  = 64, I = 32, F = 110;

    const float* x = (const float*)d_in[0];
    const float *ec_w[5], *ec_b[5], *el_w[3], *el_b[3], *dl_w[3], *dl_b[3],
                *dc_w[5], *dc_b[5], *fc_w[3], *fc_b[3];
    for (int i = 0; i < 5; ++i) { ec_w[i] = (const float*)d_in[1  + 2*i]; ec_b[i] = (const float*)d_in[2  + 2*i]; }
    for (int i = 0; i < 3; ++i) { el_w[i] = (const float*)d_in[11 + 2*i]; el_b[i] = (const float*)d_in[12 + 2*i]; }
    for (int i = 0; i < 3; ++i) { dl_w[i] = (const float*)d_in[17 + 2*i]; dl_b[i] = (const float*)d_in[18 + 2*i]; }
    for (int i = 0; i < 5; ++i) { dc_w[i] = (const float*)d_in[23 + 2*i]; dc_b[i] = (const float*)d_in[24 + 2*i]; }
    for (int i = 0; i < 3; ++i) { fc_w[i] = (const float*)d_in[33 + 2*i]; fc_b[i] = (const float*)d_in[34 + 2*i]; }

    float* out_cls = (float*)d_out;                 // [64,4]
    float* out_dec = (float*)d_out + 256;           // [2048,3,32,32]

    // --- workspace layout (256B-aligned bump allocator) ---
    char* ws = (char*)d_ws;
    size_t cur = 0;
    auto alloc = [&](size_t bytes) -> char* {
        char* p = ws + cur;
        cur += (bytes + 255) & ~(size_t)255;
        return p;
    };
    float*     BA   = (float*)alloc((size_t)BI*18*32*32*4);  // 151 MB ping
    float*     BB   = (float*)alloc((size_t)BI*18*16*16*4);  // 38 MB pong
    _Float16*  H    = (_Float16*)alloc((size_t)BI*576*2);    // f16 activations
    float*     FEAT = (float*)alloc((size_t)BI*F*4);         // [2048,110]
    float*     FD   = (float*)alloc((size_t)B*1210*4);       // [64,1210]
    _Float16*  W16[9];
    const int  WN[9]   = {576, 288, 110, 288, 576, 576, 384, 192, 4};
    const int  WNP[9]  = {576, 320, 128, 320, 576, 576, 384, 192, 64};  // mult of 64
    const int  WK[9]   = {576, 576, 288, 110, 288, 576, 1210, 384, 192};
    const int  WKP[9]  = {576, 576, 288, 128, 288, 576, 1216, 384, 192}; // mult of 32
    for (int i = 0; i < 9; ++i) W16[i] = (_Float16*)alloc((size_t)WNP[i]*WKP[i]*2);

    // --- helpers ---
    auto cast_pad = [&](const float* src, _Float16* dst, int rows, int K, int Rp, int Kp) {
        long long total = (long long)Rp * Kp;
        cast_pad_f16_kernel<<<(unsigned)CDIV(total,256), 256, 0, stream>>>(src, dst, rows, K, Rp, Kp);
    };
    auto gemm = [&](const _Float16* Aa, const _Float16* Ww, const float* bias,
                    float* C, int M, int N, int Np, int Kp, int relu) {
        int tiles = (M/16) * (Np/64);
        gemm_wmma_f16_kernel<<<(unsigned)CDIV(tiles,4), 128, 0, stream>>>(Aa, Ww, bias, C, M, N, Np, Kp, relu);
    };
    auto conv = [&](const float* in, const float* w, const float* bias, float* o,
                    int Cin, int Hin, int Cout, int Hout, int s, int relu) {
        long long total = (long long)BI*Cout*Hout*Hout;
        conv3x3_kernel<<<(unsigned)CDIV(total,256), 256, 0, stream>>>(
            in, w, bias, o, BI, Cin, Hin, Hin, Cout, Hout, Hout, s, relu);
    };
    auto convT = [&](const float* in, const float* w, const float* bias, float* o,
                     int Cin, int Hin, int Cout, int Hout, int s, int relu) {
        long long total = (long long)BI*Cout*Hout*Hout;
        convT3x3_kernel<<<(unsigned)CDIV(total,256), 256, 0, stream>>>(
            in, w, bias, o, BI, Cin, Hin, Hin, Cout, Hout, Hout, s, relu);
    };

    // --- weight precision conversion (f32 [out,in] -> f16 [Np,Kp], zero-padded) ---
    const float* wsrc[9] = {el_w[0], el_w[1], el_w[2], dl_w[0], dl_w[1], dl_w[2],
                            fc_w[0], fc_w[1], fc_w[2]};
    for (int i = 0; i < 9; ++i) cast_pad(wsrc[i], W16[i], WN[i], WK[i], WNP[i], WKP[i]);

    // --- encoder convs ---
    conv(x,  ec_w[0], ec_b[0], BA, 3, 32, 18, 32, 1, 1);
    conv(BA, ec_w[1], ec_b[1], BB, 18, 32, 18, 16, 2, 1);
    conv(BB, ec_w[2], ec_b[2], BA, 18, 16, 18, 16, 1, 1);
    conv(BA, ec_w[3], ec_b[3], BB, 18, 16,  9, 16, 1, 1);
    conv(BB, ec_w[4], ec_b[4], BA,  9, 16,  9,  8, 2, 1);   // BA = [2048,576]

    // --- encoder linears (WMMA) ---
    cast_pad(BA, H, BI, 576, BI, 576);
    gemm(H, W16[0], el_b[0], BB, BI, 576, 576, 576, 1);
    cast_pad(BB, H, BI, 576, BI, 576);
    gemm(H, W16[1], el_b[1], BA, BI, 288, 320, 576, 1);
    cast_pad(BA, H, BI, 288, BI, 288);
    gemm(H, W16[2], el_b[2], FEAT, BI, 110, 128, 288, 0);   // feat [2048,110]

    // --- decoder linears (no activations) ---
    cast_pad(FEAT, H, BI, 110, BI, 128);
    gemm(H, W16[3], dl_b[0], BA, BI, 288, 320, 128, 0);
    cast_pad(BA, H, BI, 288, BI, 288);
    gemm(H, W16[4], dl_b[1], BB, BI, 576, 576, 288, 0);
    cast_pad(BB, H, BI, 576, BI, 576);
    gemm(H, W16[5], dl_b[2], BA, BI, 576, 576, 576, 0);     // BA = [2048,9,8,8]

    // --- decoder conv-transposes ---
    convT(BA, dc_w[0], dc_b[0], BB,  9,  8,  9, 16, 2, 1);
    convT(BB, dc_w[1], dc_b[1], BA,  9, 16,  9, 16, 1, 1);
    convT(BA, dc_w[2], dc_b[2], BB,  9, 16, 18, 16, 1, 1);
    convT(BB, dc_w[3], dc_b[3], BA, 18, 16, 18, 32, 2, 1);
    convT(BA, dc_w[4], dc_b[4], out_dec, 18, 32, 3, 32, 1, 0);

    // --- KDE feature distributions ---
    kde_kernel<<<CDIV(B*F,256), 256, 0, stream>>>(FEAT, FD, B, I, F);

    // --- classifier MLP + softmax ---
    cast_pad(FD, H, B, 1210, B, 1216);
    gemm(H, W16[6], fc_b[0], BB, B, 384, 384, 1216, 1);
    cast_pad(BB, H, B, 384, B, 384);
    gemm(H, W16[7], fc_b[1], BA, B, 192, 192, 384, 1);
    cast_pad(BA, H, B, 192, B, 192);
    gemm(H, W16[8], fc_b[2], BB, B, 4, 64, 192, 0);         // logits [64,4]
    softmax4_kernel<<<CDIV(B,64), 64, 0, stream>>>(BB, out_cls, B);
}
